// ShapesReceiver_38147899523420
// MI455X (gfx1250) — matrix operand order repaired and checked by reading, not compiled, not run on hardware
//
#include <hip/hip_runtime.h>
#include <hip/hip_bf16.h>
#include <stdint.h>

// ---------------------------------------------------------------------------
// Problem constants (from reference): B=512, T=128, V=1024, E=256, H=512
// ---------------------------------------------------------------------------
#define BATCH   512
#define TSTEPS  128
#define VDIM    1024
#define EDIM    256
#define HDIM    512
#define KTOT    768          // E + H : fused [emb_t | h] contraction length
#define BT      (BATCH*TSTEPS)
#define KCHUNKS (KTOT/32)    // 24
#define NTILES  (HDIM/16)    // 32

typedef __attribute__((ext_vector_type(16))) __bf16 v16bf;
typedef __attribute__((ext_vector_type(8)))  float  v8f;

// ---------------------------------------------------------------------------
// Fragment-order storage (CDNA5 ISA 7.12.2, wave32):
// A 16x32 bf16 element (m,k): lane = m + 16*(k>>4), slot = k&15
// B 32x16 bf16 element (k,n): lane = (n&15) + 16*(k>>4),  slot = k&15
// Fragments stored [lane][slot] = 32 lanes x 16 bf16 = 1 KB, so each lane's
// WMMA operand is one contiguous 32-byte block -> 2x ds_load_b128.
// C 16x16 f32: VGPR r -> M = r + 8*(lane>=16), N = lane&15.
// ---------------------------------------------------------------------------
__device__ inline v16bf load_frag(const __bf16* base) {
  const int lane = threadIdx.x & 31;
  return *(const v16bf*)(base + lane * 16);
}

__device__ inline float fast_sigmoid(float x) {
  return 1.0f / (1.0f + __expf(-x));
}

// Async global->LDS copy, 16 bytes per lane (CDNA5 GLOBAL_LOAD_ASYNC_TO_LDS).
// Tracked with ASYNCcnt; s_wait_asynccnt 0 before the LDS data is consumed.
__device__ inline void async_copy16(unsigned lds_off, const void* gptr) {
  unsigned long long ga = (unsigned long long)(uintptr_t)gptr;
  asm volatile("global_load_async_to_lds_b128 %0, %1, off"
               :: "v"(lds_off), "v"(ga) : "memory");
}
__device__ inline void wait_async() {
  asm volatile("s_wait_asynccnt 0x0" ::: "memory");
}

// pack two fp32 -> two bf16 -> one dword LDS store (v_cvt_pk_bf16_f32)
__device__ inline void store_pk_bf16(__bf16* d, float x, float y) {
  union { unsigned u; __bf16 h[2]; } pk;
  pk.h[0] = (__bf16)x; pk.h[1] = (__bf16)y;
  *(unsigned*)d = pk.u;
}

// ---------------------------------------------------------------------------
// Prep: zero h0/c0, combine biases, build bf16 weights PRE-SWIZZLED into
// WMMA fragment order:
//   frag index = (g*KCHUNKS + (k>>5))*NTILES + (n>>4), 512 bf16 per fragment,
//   within fragment [lane][slot] per the B-operand layout above, so the step
//   kernel's weight staging is a pure 16B-block global->LDS async copy.
// ---------------------------------------------------------------------------
__global__ __launch_bounds__(256) void prep_kernel(
    const float* __restrict__ W_ih, const float* __restrict__ W_hh,
    const float* __restrict__ b_ih, const float* __restrict__ b_hh,
    float* __restrict__ h0, float* __restrict__ c0,
    float* __restrict__ bias, __bf16* __restrict__ WbT)
{
  const int idx = blockIdx.x * 256 + threadIdx.x;
  if (idx < 4 * KTOT * HDIM) {
    const int n = idx & (HDIM - 1);
    const int k = (idx >> 9) % KTOT;
    const int g = idx / (KTOT * HDIM);
    const float w = (k < EDIM)
        ? W_ih[(size_t)(g * HDIM + n) * EDIM + k]
        : W_hh[(size_t)(g * HDIM + n) * HDIM + (k - EDIM)];
    const int kk   = k & 31;
    const int lane = (n & 15) + 16 * (kk >> 4);
    const int slot = kk & 15;
    const size_t frag = ((size_t)(g * KCHUNKS + (k >> 5)) * NTILES + (n >> 4));
    WbT[(frag << 9) + lane * 16 + slot] = (__bf16)w;
  }
  if (idx < 4 * HDIM) bias[idx] = b_ih[idx] + b_hh[idx];
  if (idx < BATCH * HDIM) { h0[idx] = 0.0f; c0[idx] = 0.0f; }
}

// ---------------------------------------------------------------------------
// Phase 1: emb[B*T, E] = messages[B*T, V] @ embedding[V, E]
// Tile 128(M) x 64(N), 8 waves as 4(M)x2(N); each wave: 2x2 wmma tiles.
// HBM-bound (256 MB of messages). Double-buffered LDS, one barrier per
// K-chunk: chunk k+1 is staged while chunk k runs on the WMMA pipe.
// ---------------------------------------------------------------------------
__global__ __launch_bounds__(256) void emb_gemm_kernel(
    const float* __restrict__ msgs,   // [BT, V]
    const float* __restrict__ embt,   // [V, E]
    float* __restrict__ emb_out)      // [BT, E]
{
  __shared__ __align__(32) __bf16 Asf[2][8 * 512];  // 8 A fragments (16x32)
  __shared__ __align__(32) __bf16 Bsf[2][4 * 512];  // 4 B fragments (32x16)

  const int m0   = blockIdx.y * 128;
  const int n0   = blockIdx.x * 64;
  const int tid  = threadIdx.x;
  const int wave = tid >> 5;
  const int wm   = wave >> 1;         // 0..3
  const int wn   = wave & 1;          // 0..1

  auto stage = [&](int k0, int buf) {
    // A: 128 rows x 32 K, fp32 -> bf16, swizzled (8 pairs/thread)
#pragma unroll
    for (int i = 0; i < 8; ++i) {
      const int p  = tid + i * 256;       // pair index 0..2047
      const int r  = p >> 4;              // row 0..127
      const int c2 = (p & 15) * 2;        // k 0,2,..,30
      const float2 v = *(const float2*)&msgs[(size_t)(m0 + r) * VDIM + k0 + c2];
      const int fr   = r >> 4;
      const int m    = r & 15;
      const int lane = m + 16 * (c2 >> 4);
      const int slot = c2 & 15;
      store_pk_bf16(&Asf[buf][fr * 512 + lane * 16 + slot], v.x, v.y);
    }
    // B: 32 K x 64 N, fp32 -> bf16, swizzled (8 elems/thread)
#pragma unroll
    for (int i = 0; i < 8; ++i) {
      const int idx = tid + i * 256;
      const int r = idx >> 6, c = idx & 63;   // k, n
      const int nt   = c >> 4;
      const int lane = (c & 15) + 16 * (r >> 4);
      const int slot = r & 15;
      Bsf[buf][nt * 512 + lane * 16 + slot] =
          (__bf16)embt[(size_t)(k0 + r) * EDIM + n0 + c];
    }
  };

  v8f acc[2][2];
#pragma unroll
  for (int i = 0; i < 2; ++i)
#pragma unroll
    for (int j = 0; j < 2; ++j) acc[i][j] = v8f{};

  stage(0, 0);
  __syncthreads();

  for (int kc = 0; kc < VDIM / 32; ++kc) {
    const int cur = kc & 1;
    if (kc + 1 < VDIM / 32) stage((kc + 1) * 32, cur ^ 1);
#pragma unroll
    for (int i = 0; i < 2; ++i) {
      v16bf a = load_frag(&Asf[cur][(wm * 2 + i) * 512]);
#pragma unroll
      for (int j = 0; j < 2; ++j) {
        v16bf b = load_frag(&Bsf[cur][(wn * 2 + j) * 512]);
        acc[i][j] = __builtin_amdgcn_wmma_f32_16x16x32_bf16(
            false, a, false, b, (short)0, acc[i][j], false, false);
      }
    }
    __syncthreads();
  }

  const int lane  = tid & 31;
  const int rbase = (lane >> 4) * 8;
  const int col   = lane & 15;
#pragma unroll
  for (int i = 0; i < 2; ++i)
#pragma unroll
    for (int j = 0; j < 2; ++j) {
      const int mb = m0 + wm * 32 + i * 16 + rbase;
      const int nb = n0 + wn * 32 + j * 16 + col;
#pragma unroll
      for (int r = 0; r < 8; ++r)
        emb_out[(size_t)(mb + r) * EDIM + nb] = acc[i][j][r];
    }
}

// ---------------------------------------------------------------------------
// Phase 2: one fused LSTM timestep:
//   gates[b, g*H+n] = sum_k [emb_t | h_in][b,k] * W[g][k][n] + bias
//   c = sig(f)*c + sig(i)*tanh(g);  h_out = sig(o)*tanh(c)
// Tile 32(B) x 64(H) per WG; 8 waves as 2(M)x4(N); one 16x16 wmma tile per
// gate per wave. Weights arrive via async-to-LDS b128 copies (fragment
// order), double-buffered so chunk k+1 streams in during chunk k's WMMAs;
// s_wait_asynccnt 0 + a single barrier per chunk.
// ---------------------------------------------------------------------------
__global__ __launch_bounds__(256) void lstm_step_kernel(
    const float* __restrict__ emb,    // [B, T, E]
    const __bf16* __restrict__ WbT,   // fragment-ordered [4][24][32][512]
    const float* __restrict__ bias,   // [4H] combined
    const float* __restrict__ h_in,   // [B, H]
    float* __restrict__ c,            // [B, H] (updated in place)
    float* __restrict__ h_out,        // [B, H]
    int t)
{
  __shared__ __align__(32) __bf16 As[2][2 * 512];   // 2 A fragments (16x32)
  __shared__ __align__(32) __bf16 Bs[2][16 * 512];  // 4 gates x 4 ntiles

  const int b0   = blockIdx.y * 32;
  const int n0   = blockIdx.x * 64;
  const int tid  = threadIdx.x;
  const int wave = tid >> 5;
  const int wm   = wave >> 2;           // 0..1 (rows)
  const int wn   = wave & 3;            // 0..3 (cols)

  auto issueB = [&](int kc, int buf) {
    // 16 fragments x 1KB = 64x16B blocks each, async global->LDS
#pragma unroll
    for (int i = 0; i < 4; ++i) {
      const int chunk = tid + i * 256;        // 0..1023 (16B units)
      const int f   = chunk >> 6;             // fragment 0..15 (g*4 + ntile)
      const int w16 = chunk & 63;             // 16B block within fragment
      const int g = f >> 2, j = f & 3;
      const size_t gfrag =
          (size_t)(g * KCHUNKS + kc) * NTILES + (n0 >> 4) + j;
      const __bf16* src = WbT + (gfrag << 9) + w16 * 8;
      const unsigned lds_off =
          (unsigned)(uintptr_t)&Bs[buf][f * 512 + w16 * 8];
      async_copy16(lds_off, src);
    }
  };
  auto stageA = [&](int k0, int buf) {
    // 32 rows x 32 K of [emb_t | h_in], fp32 -> bf16 (2 pairs/thread)
#pragma unroll
    for (int i = 0; i < 2; ++i) {
      const int p  = tid + i * 256;       // pair index 0..511
      const int r  = p >> 4;              // row 0..31
      const int c2 = (p & 15) * 2;        // k 0,2,..,30
      const float* src = (k0 < EDIM)
          ? &emb[((size_t)(b0 + r) * TSTEPS + t) * EDIM + k0 + c2]
          : &h_in[(size_t)(b0 + r) * HDIM + (k0 - EDIM) + c2];
      const float2 v = *(const float2*)src;
      const int fr   = r >> 4;
      const int m    = r & 15;
      const int lane = m + 16 * (c2 >> 4);
      const int slot = c2 & 15;
      store_pk_bf16(&As[buf][fr * 512 + lane * 16 + slot], v.x, v.y);
    }
  };

  v8f acc[4];
#pragma unroll
  for (int g = 0; g < 4; ++g) acc[g] = v8f{};

  issueB(0, 0);
  stageA(0, 0);
  wait_async();
  __syncthreads();

  for (int kc = 0; kc < KCHUNKS; ++kc) {
    const int cur = kc & 1;
    if (kc + 1 < KCHUNKS) {
      issueB(kc + 1, cur ^ 1);          // weights stream during compute
      stageA((kc + 1) * 32, cur ^ 1);
    }
    v16bf a = load_frag(&As[cur][wm * 512]);
#pragma unroll
    for (int g = 0; g < 4; ++g) {
      v16bf b = load_frag(&Bs[cur][(g * 4 + wn) * 512]);
      acc[g] = __builtin_amdgcn_wmma_f32_16x16x32_bf16(
          false, a, false, b, (short)0, acc[g], false, false);
    }
    wait_async();                        // prefetch landed
    __syncthreads();                     // single barrier per chunk
  }

  // fused LSTM pointwise epilogue on the 16x16 C layout
  const int lane  = tid & 31;
  const int rbase = (lane >> 4) * 8;
  const int hcol  = n0 + wn * 16 + (lane & 15);
#pragma unroll
  for (int r = 0; r < 8; ++r) {
    const int b = b0 + wm * 16 + rbase + r;
    float iv = fast_sigmoid(acc[0][r] + bias[0 * HDIM + hcol]);
    float fv = fast_sigmoid(acc[1][r] + bias[1 * HDIM + hcol]);
    float gv = tanhf(acc[2][r] + bias[2 * HDIM + hcol]);
    float ov = fast_sigmoid(acc[3][r] + bias[3 * HDIM + hcol]);
    const size_t idx = (size_t)b * HDIM + hcol;
    const float cn = fv * c[idx] + iv * gv;
    c[idx] = cn;
    h_out[idx] = ov * tanhf(cn);
  }
}

// ---------------------------------------------------------------------------
// Host-side orchestration
// d_in: messages, embedding, W_ih, W_hh, b_ih, b_hh
// d_out: [h (B*H) | emb (B*T*E)] fp32
// ws layout: h_ping | h_pong | c | bias | WbT(bf16, fragment-ordered) ~6.3 MB
// ---------------------------------------------------------------------------
extern "C" void kernel_launch(void* const* d_in, const int* in_sizes, int n_in,
                              void* d_out, int out_size, void* d_ws, size_t ws_size,
                              hipStream_t stream) {
  (void)in_sizes; (void)n_in; (void)out_size; (void)ws_size;

  const float* msgs = (const float*)d_in[0];
  const float* embt = (const float*)d_in[1];
  const float* W_ih = (const float*)d_in[2];
  const float* W_hh = (const float*)d_in[3];
  const float* b_ih = (const float*)d_in[4];
  const float* b_hh = (const float*)d_in[5];

  float* out_h   = (float*)d_out;
  float* out_emb = out_h + (size_t)BATCH * HDIM;

  float*  h_ping = (float*)d_ws;
  float*  h_pong = h_ping + (size_t)BATCH * HDIM;
  float*  cbuf   = h_pong + (size_t)BATCH * HDIM;
  float*  bias   = cbuf   + (size_t)BATCH * HDIM;
  __bf16* WbT    = (__bf16*)(bias + 4 * HDIM);

  // 1) build fragment-ordered bf16 weights, combined bias, zero state
  {
    const int total = 4 * KTOT * HDIM;
    prep_kernel<<<(total + 255) / 256, 256, 0, stream>>>(
        W_ih, W_hh, b_ih, b_hh, h_ping, cbuf, bias, WbT);
  }

  // 2) emb = messages @ embedding  (HBM-bound big GEMM)
  emb_gemm_kernel<<<dim3(EDIM / 64, BT / 128), 256, 0, stream>>>(
      msgs, embt, out_emb);

  // 3) 128 fused LSTM steps; ping-pong h, last step writes h into d_out
  const float* hin = h_ping;
  for (int t = 0; t < TSTEPS; ++t) {
    float* ho = (t == TSTEPS - 1) ? out_h : ((t & 1) ? h_ping : h_pong);
    lstm_step_kernel<<<dim3(HDIM / 64, BATCH / 32), 256, 0, stream>>>(
        out_emb, WbT, bias, hin, cbuf, ho, t);
    hin = ho;
  }
}
